// AttentionM_67748814127283
// MI455X (gfx1250) — compile-verified, matrix operand order checked by reading, not run on hardware
//
#include <hip/hip_runtime.h>
#include <hip/hip_bf16.h>
#include <math.h>

typedef __attribute__((ext_vector_type(16))) _Float16 v16h;
typedef __attribute__((ext_vector_type(8)))  _Float16 v8h;
typedef __attribute__((ext_vector_type(8)))  float    v8f;

// ---------------- problem constants ----------------
constexpr int B_  = 4;
constexpr int S_  = 2048;
constexpr int SP_ = 2064;          // S + 16 pad (reference always pads 16)
constexpr int D_  = 1024;
constexpr int H_  = 16;
constexpr int QL_ = 16;            // q/k head size
constexpr int VL_ = 64;            // v head size
constexpr int SPV_ = 2080;         // kv length padded to 65 tiles of 32 for V

// ---------------- workspace layout (in _Float16 elements) ----------------
constexpr size_t XH_OFF   = 0;                                   // [B*SP][D]
constexpr size_t WTQ_OFF  = XH_OFF  + (size_t)B_*SP_*D_;         // [256][D]
constexpr size_t WTQC_OFF = WTQ_OFF + (size_t)256*D_;
constexpr size_t WTK_OFF  = WTQC_OFF+ (size_t)256*D_;
constexpr size_t WTKC_OFF = WTK_OFF + (size_t)256*D_;
constexpr size_t WTV_OFF  = WTKC_OFF+ (size_t)256*D_;            // [1024][D]
constexpr size_t WTVC_OFF = WTV_OFF + (size_t)1024*D_;
constexpr size_t QWS_OFF  = WTVC_OFF+ (size_t)1024*D_;           // [B][H][SP][16]
constexpr size_t KWS_OFF  = QWS_OFF + (size_t)B_*H_*SP_*QL_;
constexpr size_t VWS_OFF  = KWS_OFF + (size_t)B_*H_*SP_*QL_;     // [B][H][64][SPV]

// ---------------- conversion kernels ----------------
__global__ void cvt_x_f16(const float* __restrict__ x, _Float16* __restrict__ xh) {
  size_t i = (size_t)blockIdx.x * blockDim.x + threadIdx.x;
  size_t total = (size_t)B_ * SP_ * D_;
  if (i >= total) return;
  int d = (int)(i % D_);
  size_t t = i / D_;
  int pos = (int)(t % SP_);
  int b   = (int)(t / SP_);
  float v = (pos < S_) ? x[((size_t)b * S_ + pos) * D_ + d] : 0.0f;
  xh[i] = (_Float16)v;
}

// transpose-convert W[D][N] -> Wt[N][D] f16
__global__ void cvt_w_tr(const float* __restrict__ w, _Float16* __restrict__ wt, int N) {
  int i = blockIdx.x * blockDim.x + threadIdx.x;
  if (i >= D_ * N) return;
  int n = i % N;
  int d = i / N;
  wt[(size_t)n * D_ + d] = (_Float16)w[i];
}

// zero the padded tail [SP_, SPV_) of the transposed V workspace
__global__ void v_tail_zero(_Float16* __restrict__ vws) {
  int i = blockIdx.x * blockDim.x + threadIdx.x;
  constexpr int TAIL = SPV_ - SP_;
  if (i >= B_ * H_ * VL_ * TAIL) return;
  int p   = i % TAIL;
  int row = i / TAIL;
  vws[(size_t)row * SPV_ + SP_ + p] = (_Float16)0.0f;
}

// ---------------- fused GEMM + gated (CSS) projection ----------------
// One wave per (2 token-tiles x 2 col-tiles) = 32x32 output block, value + gate
// accumulators (8 v8f). 1.5 b128 loads per WMMA. mode: 0=Q, 1=K
// (layout [b][h][pos][16]), 2=V (transposed layout [b][h][vd][SPV]).
__global__ void __launch_bounds__(32) proj_css(
    const _Float16* __restrict__ xh,
    const _Float16* __restrict__ wt,   // value weights,  [N][D]
    const _Float16* __restrict__ wtc,  // gate weights,   [N][D]
    const float* __restrict__ bias,
    const float* __restrict__ biasc,
    _Float16* __restrict__ dst,
    int nct2,                          // # of 32-wide column tile PAIRS
    int mode)
{
  int ttp = blockIdx.x / nct2;         // token tile pair (32 tokens)
  int ctp = blockIdx.x % nct2;         // column tile pair (32 cols)
  int l  = threadIdx.x;
  int m  = l & 15;                     // A row / B column within tile
  int h2 = l >> 4;                     // which K-half this lane carries

  const v8h* xrow[2];
  const v8h* wrow[2];
  const v8h* wrowc[2];
#pragma unroll
  for (int i = 0; i < 2; ++i) {
    xrow[i]  = (const v8h*)(xh  + (size_t)((ttp * 2 + i) * 16 + m) * D_);
    wrow[i]  = (const v8h*)(wt  + (size_t)((ctp * 2 + i) * 16 + m) * D_);
    wrowc[i] = (const v8h*)(wtc + (size_t)((ctp * 2 + i) * 16 + m) * D_);
  }

  v8f cy[2][2], cg[2][2];
  v8f zero = {};
#pragma unroll
  for (int ti = 0; ti < 2; ++ti)
#pragma unroll
    for (int ci = 0; ci < 2; ++ci) { cy[ti][ci] = zero; cg[ti][ci] = zero; }

  for (int dt = 0; dt < D_ / 32; ++dt) {
    int c0 = dt * 4 + h2;        // halves [dt*32 + h2*8, +8)
    int c1 = dt * 4 + 2 + h2;    // halves [dt*32 + 16 + h2*8, +8)
    v16h a[2], by[2], bg[2];
#pragma unroll
    for (int ti = 0; ti < 2; ++ti) {
      v8h a0 = xrow[ti][c0], a1 = xrow[ti][c1];
#pragma unroll
      for (int i = 0; i < 8; ++i) { a[ti][i] = a0[i]; a[ti][8 + i] = a1[i]; }
    }
#pragma unroll
    for (int ci = 0; ci < 2; ++ci) {
      v8h y0 = wrow[ci][c0],  y1 = wrow[ci][c1];
      v8h g0 = wrowc[ci][c0], g1 = wrowc[ci][c1];
#pragma unroll
      for (int i = 0; i < 8; ++i) {
        by[ci][i] = y0[i]; by[ci][8 + i] = y1[i];
        bg[ci][i] = g0[i]; bg[ci][8 + i] = g1[i];
      }
    }
#pragma unroll
    for (int ti = 0; ti < 2; ++ti)
#pragma unroll
      for (int ci = 0; ci < 2; ++ci) {
        cy[ti][ci] = __builtin_amdgcn_wmma_f32_16x16x32_f16(false, a[ti], false, by[ci],
                                                            (short)0, cy[ti][ci], false, false);
        cg[ti][ci] = __builtin_amdgcn_wmma_f32_16x16x32_f16(false, a[ti], false, bg[ci],
                                                            (short)0, cg[ti][ci], false, false);
      }
  }

#pragma unroll
  for (int ci = 0; ci < 2; ++ci) {
    int ct  = ctp * 2 + ci;
    int col = ct * 16 + m;
    float bb = bias[col];
    float bc = biasc[col];
#pragma unroll
    for (int ti = 0; ti < 2; ++ti) {
      int tt = ttp * 2 + ti;
#pragma unroll
      for (int r = 0; r < 8; ++r) {
        int tok = tt * 16 + r + 8 * h2;    // C layout: row = r + 8*(lane>=16)
        int b   = tok / SP_;
        int pos = tok % SP_;
        float yv = cy[ti][ci][r] + bb;
        float gv = cg[ti][ci][r] + bc;
        // sigmoid gate with native rcp (v_rcp_f32) instead of IEEE divide
        float o  = yv * __builtin_amdgcn_rcpf(1.0f + __expf(-gv));
        _Float16 oh = (_Float16)o;
        if (mode < 2) {
          // Q/K: head = ct, inner dim = m
          dst[(((size_t)b * H_ + ct) * SP_ + pos) * QL_ + m] = oh;
        } else {
          // V transposed: head = ct/4, vd = (ct%4)*16 + m, pos contiguous
          int hh = ct >> 2;
          int vd = ((ct & 3) << 4) + m;
          dst[(((size_t)b * H_ + hh) * VL_ + vd) * SPV_ + pos] = oh;
        }
      }
    }
  }
}

// ---------------- flash attention ----------------
// One wave per (b, h, 2 query tiles of 16). Online softmax over 65 KV tiles of
// 32; K/V tile loads shared across the two query tiles. The softmax scale is
// folded into the Q operand; score-group validity is uniform per (t,sub)
// because SP_ is a multiple of 16, so no per-lane masking is needed.
__global__ void __launch_bounds__(32) attn_flash(
    const _Float16* __restrict__ qws,
    const _Float16* __restrict__ kws,
    const _Float16* __restrict__ vws,
    float* __restrict__ out)
{
  __shared__ __align__(16) _Float16 pbuf[2 * 16 * 32];  // P staging, C->A transpose

  int qtp = blockIdx.x & 63;           // 64 query-tile pairs (qpos < 2048)
  int bh  = blockIdx.x >> 6;           // b*H + h, 0..63
  int l   = threadIdx.x;
  int m   = l & 15;
  int h2  = l >> 4;

  // Q tiles as A operands: d=16 zero-padded to K=32, pre-scaled by 1/sqrt(64)
  const _Float16 hscale = (_Float16)0.125f;
  v16h aq[2];
#pragma unroll
  for (int q = 0; q < 2; ++q) {
    int qt = qtp * 2 + q;
    const v8h* qrow = (const v8h*)(qws + ((size_t)bh * SP_ + qt * 16 + m) * QL_);
    v8h q0 = qrow[h2];
#pragma unroll
    for (int i = 0; i < 8; ++i) {
      aq[q][i] = q0[i] * hscale;
      aq[q][8 + i] = (_Float16)0.0f;
    }
  }

  v8f zero = {};
  v8f acc[2][4];
  float mrow[2][8], lrow[2][8];
#pragma unroll
  for (int q = 0; q < 2; ++q) {
#pragma unroll
    for (int j = 0; j < 4; ++j) acc[q][j] = zero;
#pragma unroll
    for (int r = 0; r < 8; ++r) { mrow[q][r] = -INFINITY; lrow[q][r] = 0.0f; }
  }

  for (int t = 0; t < 65; ++t) {
    float p[2][2][8];                  // [qtile][sub][row]
#pragma unroll
    for (int sub = 0; sub < 2; ++sub) {
      // SP_ % 16 == 0 -> a 16-key score group is entirely valid or entirely not
      bool valid = (t * 32 + sub * 16) < SP_;   // only t=64,sub=1 is invalid
      if (valid) {
        int kp = t * 32 + sub * 16 + m;         // key position (B column) of this lane
        const v8h* krow = (const v8h*)(kws + ((size_t)bh * SP_ + kp) * QL_);
        v8h k0 = krow[h2];
        v16h bk;
#pragma unroll
        for (int i = 0; i < 8; ++i) { bk[i] = k0[i]; bk[8 + i] = (_Float16)0.0f; }
#pragma unroll
        for (int q = 0; q < 2; ++q) {
          v8f c = __builtin_amdgcn_wmma_f32_16x16x32_f16(false, aq[q], false, bk,
                                                         (short)0, zero, false, false);
#pragma unroll
          for (int r = 0; r < 8; ++r) p[q][sub][r] = c[r];
        }
      } else {
#pragma unroll
        for (int q = 0; q < 2; ++q)
#pragma unroll
          for (int r = 0; r < 8; ++r) p[q][sub][r] = -INFINITY;
      }
    }

#pragma unroll
    for (int q = 0; q < 2; ++q) {
      // tile row-max across the 16 columns (16-lane halves reduce separately)
      float tmax[8];
#pragma unroll
      for (int r = 0; r < 8; ++r) tmax[r] = fmaxf(p[q][0][r], p[q][1][r]);
#pragma unroll
      for (int off = 1; off < 16; off <<= 1)
#pragma unroll
        for (int r = 0; r < 8; ++r) tmax[r] = fmaxf(tmax[r], __shfl_xor(tmax[r], off, 32));

      float alpha[8], rsum[8];
#pragma unroll
      for (int r = 0; r < 8; ++r) {
        float mn = fmaxf(mrow[q][r], tmax[r]);
        alpha[r] = __expf(mrow[q][r] - mn);
        mrow[q][r] = mn;
        p[q][0][r] = __expf(p[q][0][r] - mn);
        p[q][1][r] = __expf(p[q][1][r] - mn);
        rsum[r] = p[q][0][r] + p[q][1][r];
      }
#pragma unroll
      for (int off = 1; off < 16; off <<= 1)
#pragma unroll
        for (int r = 0; r < 8; ++r) rsum[r] += __shfl_xor(rsum[r], off, 32);
#pragma unroll
      for (int r = 0; r < 8; ++r) lrow[q][r] = lrow[q][r] * alpha[r] + rsum[r];
#pragma unroll
      for (int j = 0; j < 4; ++j)
#pragma unroll
        for (int r = 0; r < 8; ++r) acc[q][j][r] *= alpha[r];
    }

    // transpose both P tiles (C layout) -> A layout through LDS
    __syncthreads();
#pragma unroll
    for (int q = 0; q < 2; ++q)
#pragma unroll
      for (int r = 0; r < 8; ++r) {
        int row = r + 8 * h2;
        pbuf[q * 512 + row * 32 + m]      = (_Float16)p[q][0][r];
        pbuf[q * 512 + row * 32 + 16 + m] = (_Float16)p[q][1][r];
      }
    __syncthreads();
    v16h ap[2];
#pragma unroll
    for (int q = 0; q < 2; ++q) {
      const v8h* pr = (const v8h*)(pbuf + q * 512 + m * 32);
      v8h x0 = pr[h2];
      v8h x1 = pr[2 + h2];
#pragma unroll
      for (int i = 0; i < 8; ++i) { ap[q][i] = x0[i]; ap[q][8 + i] = x1[i]; }
    }

    // P x V : V tile loads shared across both query tiles
#pragma unroll
    for (int j = 0; j < 4; ++j) {
      const v8h* vr = (const v8h*)(vws + ((size_t)bh * VL_ + j * 16 + m) * SPV_ + t * 32);
      v8h v0 = vr[h2];
      v8h v1 = vr[2 + h2];
      v16h bv;
#pragma unroll
      for (int i = 0; i < 8; ++i) { bv[i] = v0[i]; bv[8 + i] = v1[i]; }
#pragma unroll
      for (int q = 0; q < 2; ++q)
        acc[q][j] = __builtin_amdgcn_wmma_f32_16x16x32_f16(false, ap[q], false, bv,
                                                           (short)0, acc[q][j], false, false);
    }
  }

  // epilogue: ctx = acc * (1/l) via native rcp, written to [B][S][H*VL]
  int b  = bh >> 4;
  int hh = bh & 15;
#pragma unroll
  for (int q = 0; q < 2; ++q) {
    int qt = qtp * 2 + q;
    float linv[8];
#pragma unroll
    for (int r = 0; r < 8; ++r) linv[r] = __builtin_amdgcn_rcpf(lrow[q][r]);
#pragma unroll
    for (int j = 0; j < 4; ++j)
#pragma unroll
      for (int r = 0; r < 8; ++r) {
        int qpos = qt * 16 + r + 8 * h2;     // < 2048 by construction
        int col  = hh * VL_ + j * 16 + m;
        out[((size_t)b * S_ + qpos) * (H_ * VL_) + col] = acc[q][j][r] * linv[r];
      }
  }
}

// ---------------- host launcher ----------------
extern "C" void kernel_launch(void* const* d_in, const int* in_sizes, int n_in,
                              void* d_out, int out_size, void* d_ws, size_t ws_size,
                              hipStream_t stream) {
  const float* x   = (const float*)d_in[0];
  const float* Wq  = (const float*)d_in[1];
  const float* bq  = (const float*)d_in[2];
  const float* Wqc = (const float*)d_in[3];
  const float* bqc = (const float*)d_in[4];
  const float* Wk  = (const float*)d_in[5];
  const float* bk  = (const float*)d_in[6];
  const float* Wkc = (const float*)d_in[7];
  const float* bkc = (const float*)d_in[8];
  const float* Wv  = (const float*)d_in[9];
  const float* bv  = (const float*)d_in[10];
  const float* Wvc = (const float*)d_in[11];
  const float* bvc = (const float*)d_in[12];
  _Float16* ws = (_Float16*)d_ws;
  float* out = (float*)d_out;

  // x -> f16 (with zero pad rows)
  {
    size_t total = (size_t)B_ * SP_ * D_;
    cvt_x_f16<<<(unsigned)((total + 255) / 256), 256, 0, stream>>>(x, ws + XH_OFF);
  }
  // weights -> transposed f16
  cvt_w_tr<<<(D_ * 256 + 255) / 256, 256, 0, stream>>>(Wq,  ws + WTQ_OFF,  256);
  cvt_w_tr<<<(D_ * 256 + 255) / 256, 256, 0, stream>>>(Wqc, ws + WTQC_OFF, 256);
  cvt_w_tr<<<(D_ * 256 + 255) / 256, 256, 0, stream>>>(Wk,  ws + WTK_OFF,  256);
  cvt_w_tr<<<(D_ * 256 + 255) / 256, 256, 0, stream>>>(Wkc, ws + WTKC_OFF, 256);
  cvt_w_tr<<<(D_ * 1024 + 255) / 256, 256, 0, stream>>>(Wv,  ws + WTV_OFF,  1024);
  cvt_w_tr<<<(D_ * 1024 + 255) / 256, 256, 0, stream>>>(Wvc, ws + WTVC_OFF, 1024);
  v_tail_zero<<<(B_ * H_ * VL_ * (SPV_ - SP_) + 255) / 256, 256, 0, stream>>>(ws + VWS_OFF);

  // gated projections (WMMA, 32x32 register blocks)
  int nttp = (B_ * SP_) / 32;   // 258 token tile pairs
  proj_css<<<nttp * 8,  32, 0, stream>>>(ws + XH_OFF, ws + WTQ_OFF, ws + WTQC_OFF,
                                         bq, bqc, ws + QWS_OFF, 8, 0);
  proj_css<<<nttp * 8,  32, 0, stream>>>(ws + XH_OFF, ws + WTK_OFF, ws + WTKC_OFF,
                                         bk, bkc, ws + KWS_OFF, 8, 1);
  proj_css<<<nttp * 32, 32, 0, stream>>>(ws + XH_OFF, ws + WTV_OFF, ws + WTVC_OFF,
                                         bv, bvc, ws + VWS_OFF, 32, 2);

  // flash attention (WMMA + LDS transpose + online softmax, 2 q-tiles/wave)
  attn_flash<<<64 * 64, 32, 0, stream>>>(ws + QWS_OFF, ws + KWS_OFF, ws + VWS_OFF, out);

  (void)in_sizes; (void)n_in; (void)out_size; (void)ws_size;
}
